// MM_MIL_VisionTransformer_70153995813412
// MI455X (gfx1250) — matrix operand order repaired and checked by reading, not compiled, not run on hardware
//
#include <hip/hip_runtime.h>
#include <hip/hip_bf16.h>
#include <math.h>

// ---------------------------------------------------------------------------
// MI455X (gfx1250) LoFTR encoder stack.
// Compute-bound (~2 TFLOP vs microseconds of HBM traffic at 23.3 TB/s) =>
// dense bf16 WMMA path (v_wmma_f32_16x16x32_bf16, f32 accumulate) for every
// matmul; fused flash attention avoids materializing 256MB of scores.
// 32x32 wave tiles (4 WMMA/k-step, 2x fragment reuse) and
// GLOBAL_LOAD_ASYNC_TO_LDS_B128 staging for transpose-free tiles.
// ---------------------------------------------------------------------------

typedef __attribute__((ext_vector_type(16))) __bf16 v16bf;
typedef __attribute__((ext_vector_type(8)))  float  v8f;
typedef __attribute__((ext_vector_type(4)))  int    v4i_t;

#define DEV __device__ __forceinline__
#define AS1 __attribute__((address_space(1)))
#define AS3 __attribute__((address_space(3)))

#if __has_builtin(__builtin_amdgcn_global_load_async_to_lds_b128)
#define HAVE_ASYNC_LDS 1
#else
#define HAVE_ASYNC_LDS 0
#endif

// 16B global->LDS copy: async DMA path (ASYNCcnt) when available.
DEV void stage_b128(const unsigned short* g, unsigned short* l) {
#if HAVE_ASYNC_LDS
  __builtin_amdgcn_global_load_async_to_lds_b128((AS1 v4i_t*)g, (AS3 v4i_t*)l,
                                                 0, 0);
#else
  *(uint4*)l = *(const uint4*)g;
#endif
}

DEV void wait_stage() {
#if HAVE_ASYNC_LDS
  asm volatile("s_wait_asynccnt 0" ::: "memory");
#endif
}

DEV unsigned short f2bf(float f) {
  unsigned int u = __float_as_uint(f);
  unsigned int r = u + 0x7FFFu + ((u >> 16) & 1u);  // round-to-nearest-even
  return (unsigned short)(r >> 16);
}

DEV v8f zero8() {
  v8f z;
#pragma unroll
  for (int i = 0; i < 8; ++i) z[i] = 0.f;
  return z;
}

DEV v8f wmma_bf16(v16bf a, v16bf b, v8f c) {
  return __builtin_amdgcn_wmma_f32_16x16x32_bf16(false, a, false, b,
                                                 (short)0, c, false, false);
}

// A-fragment (16x32 bf16, MxK): lane&15 = row; lanes>=16 hold k+16 offset.
// VGPR j holds k-pair base 2j; +16 for hi lanes.
DEV v16bf load_afrag_row(const unsigned short* prow, int lane) {
  union { v16bf v; unsigned int u[8]; } f;
  const int hi = (lane >= 16) ? 16 : 0;
#pragma unroll
  for (int j = 0; j < 8; ++j) {
    int kb = 2 * j + hi;
    f.u[j] = *(const unsigned int*)(prow + kb);
  }
  return f.v;
}

// B-fragment (32x16 bf16, KxN): lane&15 = column n; lanes 0-15 hold k 0..15,
// lanes 16-31 hold k 16..31; VGPR j = k-pair (kh+2j, kh+2j+1).
DEV v16bf load_bfrag(const unsigned short* pcol, int lane) {
  union { v16bf v; unsigned int u[8]; } f;
  const int kh = (lane >= 16) ? 16 : 0;
#pragma unroll
  for (int j = 0; j < 8; ++j)
    f.u[j] = *(const unsigned int*)(pcol + kh + 2 * j);
  return f.v;
}

// ---------------------------------------------------------------------------
// GEMM: C[M,N] (f32 acc) = A_bf16[M,K](lda) @ B_bf16[K,N](ldb)
// 128x64 block tile, BK=32, 256 threads = 8 waves (4M x 2N), each wave owns a
// 32x32 slab = 4 WMMAs per k-step with 2x A/B fragment reuse.
// A tile staged via async-to-LDS (straight copy); B staged transposed [n][k]
// through VGPRs so fragment k-pairs stay dword-contiguous.
// EPI: 0 = store f32, 1 = store bf16, 2 = relu + store bf16
// ---------------------------------------------------------------------------
template <int EPI>
__global__ __launch_bounds__(256) void gemm_bf16_k(
    const unsigned short* __restrict__ A, int lda,
    const unsigned short* __restrict__ B, int ldb,
    void* __restrict__ Cout, int ldc, int K) {
  __shared__ __align__(16) unsigned short As[128 * 32];  // [m][k], 8KB
  __shared__ __align__(16) unsigned short BsT[64 * 32];  // [n][k], 4KB
  const int tid = threadIdx.x;
  const int lane = tid & 31;
  const int wave = tid >> 5;
  const int wm = wave & 3;   // 0..3 -> 32-row slab
  const int wn = wave >> 2;  // 0..1 -> 32-col slab
  const int row0 = blockIdx.y * 128;
  const int col0 = blockIdx.x * 64;

  v8f c00 = zero8(), c01 = zero8(), c10 = zero8(), c11 = zero8();
  const int nk = K >> 5;
  for (int kt = 0; kt < nk; ++kt) {
    // stage A tile 128x32 bf16 = 512 x b128 (async DMA to LDS)
#pragma unroll
    for (int i = 0; i < 2; ++i) {
      int d = tid + i * 256;
      int m = d >> 2, q = d & 3;  // 4 x b128 per 32-elem row
      stage_b128(A + (size_t)(row0 + m) * lda + kt * 32 + q * 8,
                 As + m * 32 + q * 8);
    }
    // stage B tile 32x64 transposed -> [n][k]
#pragma unroll
    for (int i = 0; i < 4; ++i) {
      int d = tid + i * 256;
      int k = d >> 5, nn = d & 31;
      unsigned int u =
          ((const unsigned int*)(B + (size_t)(kt * 32 + k) * ldb + col0))[nn];
      BsT[(2 * nn) * 32 + k]     = (unsigned short)(u & 0xffffu);
      BsT[(2 * nn + 1) * 32 + k] = (unsigned short)(u >> 16);
    }
    if (kt + 1 < nk) {  // gfx1250 global_prefetch_b8 of next tiles
      __builtin_prefetch(
          (const void*)(A + (size_t)(row0 + (tid >> 1)) * lda + (kt + 1) * 32),
          0, 1);
      __builtin_prefetch(
          (const void*)(B + (size_t)((kt + 1) * 32 + (tid & 31)) * ldb + col0),
          0, 1);
    }
    wait_stage();
    __syncthreads();
    v16bf a0 = load_afrag_row(As + (wm * 32 + (lane & 15)) * 32, lane);
    v16bf a1 = load_afrag_row(As + (wm * 32 + 16 + (lane & 15)) * 32, lane);
    v16bf b0 = load_bfrag(BsT + (wn * 32 + (lane & 15)) * 32, lane);
    v16bf b1 = load_bfrag(BsT + (wn * 32 + 16 + (lane & 15)) * 32, lane);
    c00 = wmma_bf16(a0, b0, c00);
    c01 = wmma_bf16(a0, b1, c01);
    c10 = wmma_bf16(a1, b0, c10);
    c11 = wmma_bf16(a1, b1, c11);
    __syncthreads();
  }

  const int r0 = row0 + wm * 32 + ((lane >= 16) ? 8 : 0);
  const int cc = col0 + wn * 32 + (lane & 15);
  if (EPI == 0) {
    float* Cf = (float*)Cout;
#pragma unroll
    for (int v = 0; v < 8; ++v) {
      Cf[(size_t)(r0 + v) * ldc + cc]           = c00[v];
      Cf[(size_t)(r0 + v) * ldc + cc + 16]      = c01[v];
      Cf[(size_t)(r0 + 16 + v) * ldc + cc]      = c10[v];
      Cf[(size_t)(r0 + 16 + v) * ldc + cc + 16] = c11[v];
    }
  } else {
    unsigned short* Cb = (unsigned short*)Cout;
#pragma unroll
    for (int v = 0; v < 8; ++v) {
      float x00 = c00[v], x01 = c01[v], x10 = c10[v], x11 = c11[v];
      if (EPI == 2) {
        x00 = fmaxf(x00, 0.f); x01 = fmaxf(x01, 0.f);
        x10 = fmaxf(x10, 0.f); x11 = fmaxf(x11, 0.f);
      }
      Cb[(size_t)(r0 + v) * ldc + cc]           = f2bf(x00);
      Cb[(size_t)(r0 + v) * ldc + cc + 16]      = f2bf(x01);
      Cb[(size_t)(r0 + 16 + v) * ldc + cc]      = f2bf(x10);
      Cb[(size_t)(r0 + 16 + v) * ldc + cc + 16] = f2bf(x11);
    }
  }
}

// ---------------------------------------------------------------------------
// Fused flash attention, H=8 heads, D=96, L=S=1024.
// Q/K/V bf16 [N*L, 768] row-major (head h = cols h*96..h*96+95).
// grid = (L/128, H, N), 256 threads; wave w owns query rows q0 = bx*128+w*16.
// K block staged via async-to-LDS; V transposed through VGPRs.
// ---------------------------------------------------------------------------
__global__ __launch_bounds__(256) void attn_k(
    const unsigned short* __restrict__ Q, const unsigned short* __restrict__ Kt,
    const unsigned short* __restrict__ V, unsigned short* __restrict__ Msg,
    float scale) {
  __shared__ __align__(16) unsigned short Kb[32 * 96];      // [s][d]
  __shared__ __align__(16) unsigned short VT[96 * 32];      // [d][s]
  __shared__ __align__(16) unsigned short Ps[8 * 16 * 32];  // per-wave P tile
  const int tid = threadIdx.x;
  const int lane = tid & 31;
  const int wave = tid >> 5;
  const int h = blockIdx.y;
  const int n = blockIdx.z;
  const int q0 = blockIdx.x * 128 + wave * 16;
  const size_t rowbase = (size_t)n * 1024;

  // Q fragments for this wave's 16 rows: 3 k-steps over D=96
  v16bf qf[3];
  {
    const unsigned short* qrow =
        Q + (rowbase + q0 + (lane & 15)) * 768 + h * 96;
#pragma unroll
    for (int ks = 0; ks < 3; ++ks) qf[ks] = load_afrag_row(qrow + ks * 32, lane);
  }

  v8f acc[6];
#pragma unroll
  for (int t = 0; t < 6; ++t) acc[t] = zero8();
  float mrow[8], lrow[8];
#pragma unroll
  for (int v = 0; v < 8; ++v) { mrow[v] = -1e30f; lrow[v] = 0.f; }
  unsigned short* Pw = Ps + wave * (16 * 32);

  for (int sb = 0; sb < 32; ++sb) {
    const int s0 = sb * 32;
    // K block: 32 rows x 96 bf16 = 384 x b128, async DMA to LDS
#pragma unroll
    for (int i = 0; i < 2; ++i) {
      int d = tid + i * 256;
      if (d < 384) {
        int s = d / 12, q = d - s * 12;  // 12 x b128 per 96-elem row
        stage_b128(Kt + (rowbase + s0 + s) * 768 + h * 96 + q * 8,
                   Kb + s * 96 + q * 8);
      }
    }
    // V block transposed -> VT[d][s] (1536 dwords through VGPRs)
#pragma unroll
    for (int i = 0; i < 6; ++i) {
      int d = tid + i * 256;
      int s = d / 48, kd = d - s * 48;
      unsigned int u =
          ((const unsigned int*)(V + (rowbase + s0 + s) * 768 + h * 96))[kd];
      int dd = kd * 2;
      VT[dd * 32 + s]       = (unsigned short)(u & 0xffffu);
      VT[(dd + 1) * 32 + s] = (unsigned short)(u >> 16);
    }
    wait_stage();
    __syncthreads();

    // scores: 16 rows x 32 source cols = 2 C-tiles; B(k=d, n=src row)
    v8f st0 = zero8(), st1 = zero8();
#pragma unroll
    for (int ks = 0; ks < 3; ++ks) {
      v16bf b0 = load_bfrag(Kb + (lane & 15) * 96 + ks * 32, lane);
      v16bf b1 = load_bfrag(Kb + ((lane & 15) + 16) * 96 + ks * 32, lane);
      st0 = wmma_bf16(qf[ks], b0, st0);
      st1 = wmma_bf16(qf[ks], b1, st1);
    }

    // online softmax; C-layout: vgpr v holds row v (lanes<16) / v+8 (lanes>=16)
#pragma unroll
    for (int v = 0; v < 8; ++v) {
      float a = st0[v] * scale;
      float b = st1[v] * scale;
      float mx = fmaxf(a, b);
#pragma unroll
      for (int off = 8; off >= 1; off >>= 1)
        mx = fmaxf(mx, __shfl_xor(mx, off, 32));
      float mnew = fmaxf(mrow[v], mx);
      float sc = __expf(mrow[v] - mnew);
      float p0 = __expf(a - mnew);
      float p1 = __expf(b - mnew);
      float rs = p0 + p1;
#pragma unroll
      for (int off = 8; off >= 1; off >>= 1) rs += __shfl_xor(rs, off, 32);
      lrow[v] = lrow[v] * sc + rs;
      mrow[v] = mnew;
#pragma unroll
      for (int t = 0; t < 6; ++t) acc[t][v] *= sc;
      int r = v + ((lane >= 16) ? 8 : 0);
      int cc = lane & 15;
      Pw[r * 32 + cc]      = f2bf(p0);
      Pw[r * 32 + cc + 16] = f2bf(p1);
    }
    asm volatile("s_wait_dscnt 0" ::: "memory");  // wave-private LDS bounce

    // P @ V_blk : A = P (16x32), B(k=src row, n=dim) from VT, 6 output tiles
    v16bf pa = load_afrag_row(Pw + (lane & 15) * 32, lane);
#pragma unroll
    for (int t = 0; t < 6; ++t) {
      v16bf bv = load_bfrag(VT + (t * 16 + (lane & 15)) * 32, lane);
      acc[t] = wmma_bf16(pa, bv, acc[t]);
    }
    __syncthreads();
  }

  // normalize and store bf16 msg
#pragma unroll
  for (int v = 0; v < 8; ++v) {
    float inv = 1.f / lrow[v];
    int r = q0 + v + ((lane >= 16) ? 8 : 0);
    unsigned short* orow = Msg + (rowbase + r) * 768 + h * 96 + (lane & 15);
#pragma unroll
    for (int t = 0; t < 6; ++t) orow[t * 16] = f2bf(acc[t][v] * inv);
  }
}

// ---------------------------------------------------------------------------
// LayerNorm kernels: one wave per 768-wide row.
// ---------------------------------------------------------------------------
__global__ __launch_bounds__(256) void ln_store_bf16_k(
    const float* __restrict__ in, const float* __restrict__ g,
    const float* __restrict__ b, unsigned short* __restrict__ out, int ldo,
    int coloff) {
  const int lane = threadIdx.x & 31;
  const int row = (int)((blockIdx.x * 256 + threadIdx.x) >> 5);
  const float* x = in + (size_t)row * 768;
  float s = 0.f, ss = 0.f;
  for (int c = lane; c < 768; c += 32) { float v = x[c]; s += v; ss += v * v; }
#pragma unroll
  for (int off = 16; off >= 1; off >>= 1) {
    s += __shfl_xor(s, off, 32);
    ss += __shfl_xor(ss, off, 32);
  }
  float mean = s * (1.f / 768.f);
  float var = ss * (1.f / 768.f) - mean * mean;
  float rstd = rsqrtf(var + 1e-5f);
  unsigned short* o = out + (size_t)row * ldo + coloff;
  for (int c = lane; c < 768; c += 32)
    o[c] = f2bf((x[c] - mean) * rstd * g[c] + b[c]);
}

__global__ __launch_bounds__(256) void ln_residual_k(
    const float* __restrict__ in, const float* __restrict__ g,
    const float* __restrict__ b, float* __restrict__ xio) {
  const int lane = threadIdx.x & 31;
  const int row = (int)((blockIdx.x * 256 + threadIdx.x) >> 5);
  const float* x = in + (size_t)row * 768;
  float s = 0.f, ss = 0.f;
  for (int c = lane; c < 768; c += 32) { float v = x[c]; s += v; ss += v * v; }
#pragma unroll
  for (int off = 16; off >= 1; off >>= 1) {
    s += __shfl_xor(s, off, 32);
    ss += __shfl_xor(ss, off, 32);
  }
  float mean = s * (1.f / 768.f);
  float var = ss * (1.f / 768.f) - mean * mean;
  float rstd = rsqrtf(var + 1e-5f);
  float* o = xio + (size_t)row * 768;
  for (int c = lane; c < 768; c += 32)
    o[c] = o[c] + (x[c] - mean) * rstd * g[c] + b[c];
}

// f32 -> bf16, optional output stride (ldo) to write into the concat buffer.
__global__ void cvt_bf16_k(const float* __restrict__ in,
                           unsigned short* __restrict__ out, long total,
                           int ncols, int ldo) {
  long i = (long)blockIdx.x * blockDim.x + threadIdx.x;
  long stride = (long)gridDim.x * blockDim.x;
  for (; i < total; i += stride) {
    long r = i / ncols;
    int c = (int)(i - r * ncols);
    out[r * (long)ldo + c] = f2bf(in[i]);
  }
}

// ---------------------------------------------------------------------------
extern "C" void kernel_launch(void* const* d_in, const int* in_sizes, int n_in,
                              void* d_out, int out_size, void* d_ws,
                              size_t ws_size, hipStream_t stream) {
  (void)in_sizes; (void)n_in; (void)out_size; (void)ws_size;
  const int Nb = 8, L = 1024, C = 768, C2 = 1536, H = 8, NL = 8;
  const int M = Nb * L;  // 8192 rows
  const float scale = 0.1020620726159658f;  // 1/sqrt(96)

  const float* feat0 = (const float*)d_in[0];
  const float* feat1 = (const float*)d_in[1];
  const float* Wq = (const float*)d_in[2];
  const float* Wk = (const float*)d_in[3];
  const float* Wv = (const float*)d_in[4];
  const float* Wm = (const float*)d_in[5];
  const float* W1 = (const float*)d_in[6];
  const float* W2 = (const float*)d_in[7];
  const float* g1 = (const float*)d_in[8];
  const float* b1 = (const float*)d_in[9];
  const float* g2 = (const float*)d_in[10];
  const float* b2 = (const float*)d_in[11];

  // running feat0/feat1 live in the two halves of d_out ([2,N,L,C])
  float* f0 = (float*)d_out;
  float* f1 = f0 + (size_t)M * C;
  (void)hipMemcpyAsync(f0, feat0, (size_t)M * C * 4, hipMemcpyDeviceToDevice,
                       stream);
  (void)hipMemcpyAsync(f1, feat1, (size_t)M * C * 4, hipMemcpyDeviceToDevice,
                       stream);

  // workspace carve
  char* pw = (char*)d_ws;
  auto carve = [&](size_t bytes) -> void* {
    void* r = (void*)pw;
    pw += (bytes + 255) & ~(size_t)255;
    return r;
  };
  unsigned short* wq_bf = (unsigned short*)carve((size_t)C * C * 2);
  unsigned short* wk_bf = (unsigned short*)carve((size_t)C * C * 2);
  unsigned short* wv_bf = (unsigned short*)carve((size_t)C * C * 2);
  unsigned short* wm_bf = (unsigned short*)carve((size_t)C * C * 2);
  unsigned short* w1_bf = (unsigned short*)carve((size_t)C2 * C2 * 2);
  unsigned short* w2_bf = (unsigned short*)carve((size_t)C2 * C * 2);
  unsigned short* xcat  = (unsigned short*)carve((size_t)M * C2 * 2);  // [x|LN(msg)]
  unsigned short* sbf   = (unsigned short*)carve((size_t)M * C * 2);
  unsigned short* qbf   = (unsigned short*)carve((size_t)M * C * 2);
  unsigned short* kbf   = (unsigned short*)carve((size_t)M * C * 2);
  unsigned short* vbf   = (unsigned short*)carve((size_t)M * C * 2);
  unsigned short* msgbf = (unsigned short*)carve((size_t)M * C * 2);
  unsigned short* hbf   = (unsigned short*)carve((size_t)M * C2 * 2);
  float* tmpf           = (float*)carve((size_t)M * C * 4);

  const dim3 blk(256);
  auto cvt = [&](const float* in, unsigned short* out, long total, int ncols,
                 int ldo) {
    long gb = (total + 255) / 256;
    if (gb > 4096) gb = 4096;
    cvt_bf16_k<<<(unsigned)gb, blk, 0, stream>>>(in, out, total, ncols, ldo);
  };

  auto enc = [&](float* x, const float* src, int i) {
    // bf16 copies of activations (x goes into cols 0..C-1 of concat buffer)
    cvt(x, xcat, (long)M * C, C, C2);
    cvt(src, sbf, (long)M * C, C, C);
    // projections
    gemm_bf16_k<1><<<dim3(C / 64, M / 128), blk, 0, stream>>>(
        xcat, C2, wq_bf, C, (void*)qbf, C, C);
    gemm_bf16_k<1><<<dim3(C / 64, M / 128), blk, 0, stream>>>(
        sbf, C, wk_bf, C, (void*)kbf, C, C);
    gemm_bf16_k<1><<<dim3(C / 64, M / 128), blk, 0, stream>>>(
        sbf, C, wv_bf, C, (void*)vbf, C, C);
    // fused flash attention -> msg (bf16)
    attn_k<<<dim3(L / 128, H, Nb), blk, 0, stream>>>(qbf, kbf, vbf, msgbf,
                                                     scale);
    // msg @ Wm (f32) -> LN1 -> concat[:, C:2C] (bf16)
    gemm_bf16_k<0><<<dim3(C / 64, M / 128), blk, 0, stream>>>(
        msgbf, C, wm_bf, C, (void*)tmpf, C, C);
    ln_store_bf16_k<<<M / 8, blk, 0, stream>>>(tmpf, g1 + (size_t)i * C,
                                               b1 + (size_t)i * C, xcat, C2, C);
    // MLP: relu(cat @ W1) -> h (bf16); h @ W2 -> tmpf (f32)
    gemm_bf16_k<2><<<dim3(C2 / 64, M / 128), blk, 0, stream>>>(
        xcat, C2, w1_bf, C2, (void*)hbf, C2, C2);
    gemm_bf16_k<0><<<dim3(C / 64, M / 128), blk, 0, stream>>>(
        hbf, C2, w2_bf, C, (void*)tmpf, C, C2);
    // x += LN2(tmpf)
    ln_residual_k<<<M / 8, blk, 0, stream>>>(tmpf, g2 + (size_t)i * C,
                                             b2 + (size_t)i * C, x);
  };

  for (int i = 0; i < NL; ++i) {
    // per-layer weight conversion (reused by both encoder applications)
    cvt(Wq + (size_t)i * C * C, wq_bf, (long)C * C, C, C);
    cvt(Wk + (size_t)i * C * C, wk_bf, (long)C * C, C, C);
    cvt(Wv + (size_t)i * C * C, wv_bf, (long)C * C, C, C);
    cvt(Wm + (size_t)i * C * C, wm_bf, (long)C * C, C, C);
    cvt(W1 + (size_t)i * C2 * C2, w1_bf, (long)C2 * C2, C2, C2);
    cvt(W2 + (size_t)i * C2 * C, w2_bf, (long)C2 * C, C, C);
    if ((i & 1) == 0) {  // self
      enc(f0, f0, i);
      enc(f1, f1, i);
    } else {  // cross (f0 updated before feeding f1's layer, per reference)
      enc(f0, f1, i);
      enc(f1, f0, i);
    }
  }
}
